// TAD_13005160972788
// MI455X (gfx1250) — matrix operand order repaired and checked by reading, not compiled, not run on hardware
//
#include <hip/hip_runtime.h>
#include <math.h>

// ---------------- problem constants ----------------
constexpr int NV = 4096;   // nodes
constexpr int DV = 64;     // embedding dim

typedef __attribute__((ext_vector_type(16))) _Float16 v16h;
typedef __attribute__((ext_vector_type(8)))  float    v8f;

// ---- CDNA5 WMMA 16x16x32 f16 operand builders (ISA 7.12.2 layouts) ----
// A operand: lane l -> row m=l&15, half=l>>4; elem e<8 : K=half*8+e,
//            e>=8 : K=16+half*8+(e-8)
__device__ __forceinline__ v16h mk_a_f16(const _Float16* row, int kc, int half) {
  v16h a;
#pragma unroll
  for (int e = 0; e < 8; ++e) a[e]   = row[kc + half * 8 + e];
#pragma unroll
  for (int e = 0; e < 8; ++e) a[8+e] = row[kc + 16 + half * 8 + e];
  return a;
}
// B operand: lane l -> col n=l&15, half=l>>4; elem e<8 : K=half*8+e,
//            e>=8 : K=16+half*8+(e-8)
__device__ __forceinline__ v16h mk_b_f16(const _Float16* row, int kc, int half) {
  v16h b;
#pragma unroll
  for (int e = 0; e < 16; ++e) b[e] = row[kc + ((e & 8) << 1) + half * 8 + (e & 7)];
  return b;
}
__device__ __forceinline__ v8f wmma16(v16h a, v16h b, v8f c) {
  return __builtin_amdgcn_wmma_f32_16x16x32_f16(false, a, false, b,
                                                (short)0, c, false, false);
}

// ---------------- small init ----------------
__global__ void k_init(float* xbar, unsigned* enmax) {
  int t = threadIdx.x;
  if (t < 64) xbar[t] = 0.f;
  if (t == 64) *enmax = 0u;
}

// ---------------- per-node stats: sq, en, xn(f16), B^T=[x|x^2]^T(f16), mean ----
__global__ void k_node_stats(const float* __restrict__ x,
                             _Float16* __restrict__ xn16,
                             _Float16* __restrict__ bt16,
                             float* __restrict__ sq, float* __restrict__ en,
                             float* __restrict__ xbar, unsigned* __restrict__ enmax) {
  int tid = threadIdx.x, wave = tid >> 5, lane = tid & 31;
  int i = blockIdx.x * 8 + wave;
  float x0 = x[(size_t)i * DV + lane];
  float x1 = x[(size_t)i * DV + lane + 32];
  float s = x0 * x0 + x1 * x1;
#pragma unroll
  for (int o = 16; o > 0; o >>= 1) s += __shfl_xor(s, o, 32);
  float nrm = sqrtf(s);
  float rn = 1.0f / fmaxf(nrm, 1e-12f);
  xn16[(size_t)i * DV + lane]      = (_Float16)(x0 * rn);
  xn16[(size_t)i * DV + lane + 32] = (_Float16)(x1 * rn);
  bt16[(size_t)lane * NV + i]        = (_Float16)x0;
  bt16[(size_t)(lane + 32) * NV + i] = (_Float16)x1;
  bt16[(size_t)(64 + lane) * NV + i] = (_Float16)(x0 * x0);
  bt16[(size_t)(96 + lane) * NV + i] = (_Float16)(x1 * x1);
  atomicAdd(&xbar[lane], x0);
  atomicAdd(&xbar[lane + 32], x1);
  if (lane == 0) {
    sq[i] = s; en[i] = nrm;
    atomicMax(enmax, __float_as_uint(nrm));   // en >= 0 -> uint order == float order
  }
}

// ---------------- row sums (deg from A, W_i from W) ----------------
__global__ void k_rowsum_f32(const float* __restrict__ M, float* __restrict__ out) {
  int tid = threadIdx.x, wave = tid >> 5, lane = tid & 31;
  int i = blockIdx.x * 8 + wave;
  const float* row = M + (size_t)i * NV;
  float s = 0.f;
  for (int j = lane; j < NV; j += 32) s += row[j];
#pragma unroll
  for (int o = 16; o > 0; o >>= 1) s += __shfl_xor(s, o, 32);
  if (lane == 0) out[i] = s;
}
__global__ void k_rowsum_f16(const _Float16* __restrict__ M, float* __restrict__ out) {
  int tid = threadIdx.x, wave = tid >> 5, lane = tid & 31;
  int i = blockIdx.x * 8 + wave;
  const _Float16* row = M + (size_t)i * NV;
  float s = 0.f;
  for (int j = lane; j < NV; j += 32) s += (float)row[j];
#pragma unroll
  for (int o = 16; o > 0; o >>= 1) s += __shfl_xor(s, o, 32);
  if (lane == 0) out[i] = s;
}

// ---------------- S = xn @ xn^T (f16 out), 32x32 block per wave (2x2 tiles) ----
__global__ __launch_bounds__(256) void k_gemm_s(const _Float16* __restrict__ xn16,
                                                _Float16* __restrict__ s16) {
  int tid = threadIdx.x, wave = tid >> 5, lane = tid & 31;
  int t = blockIdx.x * 8 + wave;          // (NV/32)^2 = 16384 superblocks
  int si = t >> 7, sj = t & 127;
  int m = lane & 15, half = lane >> 4;
  const _Float16* ar0 = xn16 + (size_t)(si * 32 + m) * DV;
  const _Float16* ar1 = xn16 + (size_t)(si * 32 + 16 + m) * DV;
  const _Float16* br0 = xn16 + (size_t)(sj * 32 + m) * DV;       // B = xn^T
  const _Float16* br1 = xn16 + (size_t)(sj * 32 + 16 + m) * DV;
  v8f c00 = {}, c01 = {}, c10 = {}, c11 = {};
#pragma unroll
  for (int kc = 0; kc < DV; kc += 32) {
    v16h a0 = mk_a_f16(ar0, kc, half);
    v16h a1 = mk_a_f16(ar1, kc, half);
    v16h b0 = mk_b_f16(br0, kc, half);
    v16h b1 = mk_b_f16(br1, kc, half);
    c00 = wmma16(a0, b0, c00);
    c01 = wmma16(a0, b1, c01);
    c10 = wmma16(a1, b0, c10);
    c11 = wmma16(a1, b1, c11);
  }
#pragma unroll
  for (int r = 0; r < 8; ++r) {
    int mo = half * 8 + r;                // C layout: VGPR r, half-wave selects M block
    size_t r0 = (size_t)(si * 32 + mo) * NV;
    size_t r1 = (size_t)(si * 32 + 16 + mo) * NV;
    s16[r0 + sj * 32 + m]      = (_Float16)c00[r];
    s16[r0 + sj * 32 + 16 + m] = (_Float16)c01[r];
    s16[r1 + sj * 32 + m]      = (_Float16)c10[r];
    s16[r1 + sj * 32 + 16 + m] = (_Float16)c11[r];
  }
}

// ---------------- C1 = A @ [x | x^2]; one M-tile x all 8 N-tiles per wave ----
__global__ __launch_bounds__(256) void k_gemm_axb(const float* __restrict__ A,
                                                  const _Float16* __restrict__ bt16,
                                                  float* __restrict__ C1) {
  int tid = threadIdx.x, wave = tid >> 5, lane = tid & 31;
  int mt = blockIdx.x * 8 + wave;         // 256 M-tiles total
  int m = lane & 15, half = lane >> 4;
  const float* arow = A + (size_t)(mt * 16 + m) * NV;
  const _Float16* bbase = bt16 + (size_t)m * NV;
  v8f acc[8];
#pragma unroll
  for (int nt = 0; nt < 8; ++nt) acc[nt] = (v8f){};
  for (int kc = 0; kc < NV; kc += 32) {
    __builtin_prefetch(arow + kc + 256, 0, 1);      // stream A one chunk group ahead
    v16h a;
#pragma unroll
    for (int e = 0; e < 8; ++e) a[e]   = (_Float16)arow[kc + half * 8 + e];
#pragma unroll
    for (int e = 0; e < 8; ++e) a[8+e] = (_Float16)arow[kc + 16 + half * 8 + e];
#pragma unroll
    for (int nt = 0; nt < 8; ++nt) {
      v16h b = mk_b_f16(bbase + (size_t)nt * 16 * NV, kc, half);
      acc[nt] = wmma16(a, b, acc[nt]);              // A reused across all 8 N-tiles
    }
  }
#pragma unroll
  for (int nt = 0; nt < 8; ++nt)
#pragma unroll
    for (int r = 0; r < 8; ++r) {
      int mo = half * 8 + r;
      C1[(size_t)(mt * 16 + mo) * 128 + nt * 16 + m] = acc[nt][r];
    }
}

// ---------------- W_ij = A_ij / dist_ij  (f16) ----------------
__global__ void k_wmat(const float* __restrict__ A, const _Float16* __restrict__ s16,
                       const float* __restrict__ sq, const float* __restrict__ en,
                       _Float16* __restrict__ w16) {
  size_t idx = (size_t)blockIdx.x * 256 + threadIdx.x;
  int i = (int)(idx >> 12), j = (int)(idx & 4095);
  float a = A[idx];
  float w = 0.f;
  if (a > 0.f) {
    float g  = en[i] * en[j] * (float)s16[idx];
    float d2 = fmaxf(sq[i] + sq[j] - 2.f * g, 1e-12f);
    w = rsqrtf(d2);                        // 1/nrm, nrm >= 1e-6 so no extra clamp
  }
  w16[idx] = (_Float16)w;
}

// ---------------- V = W @ x; one M-tile x all 4 N-tiles per wave ----------------
__global__ __launch_bounds__(256) void k_gemm_v(const _Float16* __restrict__ w16,
                                                const _Float16* __restrict__ bt16,
                                                float* __restrict__ V) {
  int tid = threadIdx.x, wave = tid >> 5, lane = tid & 31;
  int mt = blockIdx.x * 8 + wave;         // 256 M-tiles total
  int m = lane & 15, half = lane >> 4;
  const _Float16* arow = w16 + (size_t)(mt * 16 + m) * NV;
  const _Float16* bbase = bt16 + (size_t)m * NV;    // rows 0..63 of B^T = x^T
  v8f acc[4];
#pragma unroll
  for (int nt = 0; nt < 4; ++nt) acc[nt] = (v8f){};
  for (int kc = 0; kc < NV; kc += 32) {
    __builtin_prefetch(arow + kc + 512, 0, 1);      // stream W ahead (read exactly once)
    v16h a = mk_a_f16(arow, kc, half);
#pragma unroll
    for (int nt = 0; nt < 4; ++nt) {
      v16h b = mk_b_f16(bbase + (size_t)nt * 16 * NV, kc, half);
      acc[nt] = wmma16(a, b, acc[nt]);
    }
  }
#pragma unroll
  for (int nt = 0; nt < 4; ++nt)
#pragma unroll
    for (int r = 0; r < 8; ++r) {
      int mo = half * 8 + r;
      V[(size_t)(mt * 16 + mo) * DV + nt * 16 + m] = acc[nt][r];
    }
}

// ---------------- block sum helper ----------------
__device__ __forceinline__ float block_sum(float v, float* red, int tid) {
  int lane = tid & 31, wave = tid >> 5;
#pragma unroll
  for (int o = 16; o > 0; o >>= 1) v += __shfl_xor(v, o, 32);
  __syncthreads();
  if (lane == 0) red[wave] = v;
  __syncthreads();
  float r = 0.f;
#pragma unroll
  for (int w = 0; w < 8; ++w) r += red[w];
  return r;
}

// ---------------- per-row: entropy, info-flow, 5-NN indices ----------------
__global__ __launch_bounds__(256) void k_rowpass(const float* __restrict__ A,
                                                 const _Float16* __restrict__ s16,
                                                 const float* __restrict__ sq,
                                                 const float* __restrict__ en,
                                                 const float* __restrict__ deg,
                                                 int* __restrict__ nnidx,
                                                 float* __restrict__ out_entropy,
                                                 float* __restrict__ out_iflow) {
  __shared__ float sh_s[NV];            // 16 KB
  __shared__ unsigned char sh_m[NV];    //  4 KB
  __shared__ float red[8];
  __shared__ float cd[256 * 5];         //  5 KB
  __shared__ int   ci[256 * 5];         //  5 KB
  int i = blockIdx.x;
  int tid = threadIdx.x, lane = tid & 31, wave = tid >> 5;
  float eni = en[i], sqi = sq[i], degi = deg[i];

  float bd[5]; int bj[5];
#pragma unroll
  for (int t = 0; t < 5; ++t) { bd[t] = 3.4e38f; bj[t] = -1; }
  float lmax = -3.4e38f;

  for (int j = tid; j < NV; j += 256) {
    float s = (float)s16[(size_t)i * NV + j];
    float a = A[(size_t)i * NV + j];
    int msk = a > 0.f;
    sh_s[j] = s; sh_m[j] = (unsigned char)msk;
    if (msk && s > lmax) lmax = s;
    if (j != i) {
      float d2 = sqi + sq[j] - 2.f * eni * en[j] * s;   // monotone in dist
      if (d2 < bd[4]) {
        int t = 4;
        while (t > 0 && d2 < bd[t-1]) { bd[t] = bd[t-1]; bj[t] = bj[t-1]; --t; }
        bd[t] = d2; bj[t] = j;
      }
    }
  }
  // block max
#pragma unroll
  for (int o = 16; o > 0; o >>= 1) lmax = fmaxf(lmax, __shfl_xor(lmax, o, 32));
  __syncthreads();
  if (lane == 0) red[wave] = lmax;
  __syncthreads();
  float smax = red[0];
#pragma unroll
  for (int w = 1; w < 8; ++w) smax = fmaxf(smax, red[w]);

  // masked sum of exp and sum of s
  float se = 0.f, ss = 0.f;
  for (int j = tid; j < NV; j += 256)
    if (sh_m[j]) { float s = sh_s[j]; se += __expf(s - smax); ss += s; }
  float Z  = block_sum(se, red, tid);
  float ms = block_sum(ss, red, tid) / degi;

  float invZ = 1.f / Z, ent = 0.f, iv = 0.f;
  for (int j = tid; j < NV; j += 256)
    if (sh_m[j]) {
      float s = sh_s[j];
      float p = __expf(s - smax) * invZ;
      ent += p * __logf(p + 1e-8f);
      float d = s - ms;
      iv += d * d;
    }
  ent = block_sum(ent, red, tid);
  iv  = block_sum(iv, red, tid);
  if (tid == 0) {
    out_entropy[i] = -ent;
    out_iflow[i]   = iv / (degi - 1.f);
  }

  // merge per-thread top-5 candidates
#pragma unroll
  for (int t = 0; t < 5; ++t) { cd[tid * 5 + t] = bd[t]; ci[tid * 5 + t] = bj[t]; }
  __syncthreads();
  if (tid == 0) {
    float gb[5]; int gi[5];
#pragma unroll
    for (int t = 0; t < 5; ++t) { gb[t] = 3.4e38f; gi[t] = -1; }
    for (int q = 0; q < 256 * 5; ++q) {
      float d2 = cd[q]; int jj = ci[q];
      if (d2 < gb[4]) {
        int t = 4;
        while (t > 0 && d2 < gb[t-1]) { gb[t] = gb[t-1]; gi[t] = gi[t-1]; --t; }
        gb[t] = d2; gi[t] = jj;
      }
    }
#pragma unroll
    for (int t = 0; t < 5; ++t) nnidx[(size_t)i * 5 + t] = gi[t];
  }
}

// ---------------- finalize: one wave per node, 5 of the 7 outputs ----------------
__global__ void k_finalize(const float* __restrict__ x, const float* __restrict__ C1,
                           const float* __restrict__ V, const float* __restrict__ sq,
                           const float* __restrict__ en, const float* __restrict__ deg,
                           const float* __restrict__ Wrow,
                           const float* __restrict__ xbar_sum,
                           const unsigned* __restrict__ enmax_bits,
                           const int* __restrict__ nnidx, float* __restrict__ out) {
  int tid = threadIdx.x, wave = tid >> 5, lane = tid & 31;
  int i = blockIdx.x * 8 + wave;
  float x0 = x[(size_t)i * DV + lane],       x1 = x[(size_t)i * DV + lane + 32];
  float nb0 = C1[(size_t)i * 128 + lane],    nb1 = C1[(size_t)i * 128 + lane + 32];
  float e20 = C1[(size_t)i * 128 + 64 + lane], e21 = C1[(size_t)i * 128 + 96 + lane];
  float v0 = V[(size_t)i * DV + lane],       v1 = V[(size_t)i * DV + lane + 32];
  float degi = deg[i];
  float xb0 = xbar_sum[lane] * (1.f / NV), xb1 = xbar_sum[lane + 32] * (1.f / NV);
  float lm0 = 0.f, lm1 = 0.f;
#pragma unroll
  for (int t = 0; t < 5; ++t) {
    int j = nnidx[(size_t)i * 5 + t];
    lm0 += x[(size_t)j * DV + lane];
    lm1 += x[(size_t)j * DV + lane + 32];
  }
  lm0 *= 0.2f; lm1 *= 0.2f;
  float dot_xnb = x0 * nb0 + x1 * nb1;
  float varsum  = (e20 - nb0 * nb0 / degi) + (e21 - nb1 * nb1 / degi);
  float g0 = x0 - xb0, g1 = x1 - xb1; float gdev2 = g0 * g0 + g1 * g1;
  float l0 = x0 - lm0, l1 = x1 - lm1; float ldev2 = l0 * l0 + l1 * l1;
  float vdotx = v0 * x0 + v1 * x1;    float vn2 = v0 * v0 + v1 * v1;
#pragma unroll
  for (int o = 16; o > 0; o >>= 1) {
    dot_xnb += __shfl_xor(dot_xnb, o, 32);
    varsum  += __shfl_xor(varsum,  o, 32);
    gdev2   += __shfl_xor(gdev2,   o, 32);
    ldev2   += __shfl_xor(ldev2,   o, 32);
    vdotx   += __shfl_xor(vdotx,   o, 32);
    vn2     += __shfl_xor(vn2,     o, 32);
  }
  if (lane == 0) {
    float sqi = sq[i];
    float te = sqi + (-dot_xnb) / (degi + 1e-8f);
    float eq = 0.6f * sqrtf(gdev2) + 0.4f * sqrtf(ldev2);
    float enmax = __uint_as_float(*enmax_bits);
    float sc = 0.5f * degi / (float)(NV - 1) + 0.5f * en[i] / enmax;
    float st = 1.f / (1.f + varsum * (1.f / DV) / (degi - 1.f));
    float Wi = Wrow[i];
    float s2 = vn2 - 2.f * Wi * vdotx + Wi * Wi * sqi;  // ||sum u||^2
    float mc = (s2 - degi) / (degi * (degi - 1.f));     // ssq == deg (units)
    out[0 * NV + i] = te;
    out[2 * NV + i] = eq;
    out[3 * NV + i] = sc;
    out[5 * NV + i] = st;
    out[6 * NV + i] = 1.f - mc;
  }
}

extern "C" void kernel_launch(void* const* d_in, const int* in_sizes, int n_in,
                              void* d_out, int out_size, void* d_ws, size_t ws_size,
                              hipStream_t stream) {
  (void)in_sizes; (void)n_in; (void)out_size; (void)ws_size;
  const float* x = (const float*)d_in[0];
  const float* A = (const float*)d_in[1];
  float* out = (float*)d_out;

  char* w = (char*)d_ws;
  size_t off = 0;
  auto take = [&](size_t bytes) -> void* {
    void* p = w + off;
    off += (bytes + 255) & ~(size_t)255;
    return p;
  };
  _Float16* s16  = (_Float16*)take((size_t)NV * NV * 2);   // 32 MB cosine sims
  _Float16* w16  = (_Float16*)take((size_t)NV * NV * 2);   // 32 MB curvature weights
  _Float16* xn16 = (_Float16*)take((size_t)NV * DV * 2);
  _Float16* bt16 = (_Float16*)take((size_t)128 * NV * 2);  // [x|x^2]^T
  float* C1   = (float*)take((size_t)NV * 128 * 4);        // [nb | ex2]
  float* V    = (float*)take((size_t)NV * DV * 4);
  float* deg  = (float*)take((size_t)NV * 4);
  float* sq   = (float*)take((size_t)NV * 4);
  float* en   = (float*)take((size_t)NV * 4);
  float* Wrow = (float*)take((size_t)NV * 4);
  float* xbar = (float*)take(64 * 4);
  unsigned* enmax = (unsigned*)take(4);
  int* nnidx = (int*)take((size_t)NV * 5 * 4);

  k_init<<<1, 128, 0, stream>>>(xbar, enmax);
  k_node_stats<<<NV / 8, 256, 0, stream>>>(x, xn16, bt16, sq, en, xbar, enmax);
  k_rowsum_f32<<<NV / 8, 256, 0, stream>>>(A, deg);
  k_gemm_s<<<(NV / 32) * (NV / 32) / 8, 256, 0, stream>>>(xn16, s16);
  k_gemm_axb<<<(NV / 16) / 8, 256, 0, stream>>>(A, bt16, C1);
  k_wmat<<<(NV / 256) * NV, 256, 0, stream>>>(A, s16, sq, en, w16);
  k_rowsum_f16<<<NV / 8, 256, 0, stream>>>(w16, Wrow);
  k_gemm_v<<<(NV / 16) / 8, 256, 0, stream>>>(w16, bt16, V);
  k_rowpass<<<NV, 256, 0, stream>>>(A, s16, sq, en, deg, nnidx,
                                    out + 1 * NV, out + 4 * NV);
  k_finalize<<<NV / 8, 256, 0, stream>>>(x, C1, V, sq, en, deg, Wrow,
                                         xbar, enmax, nnidx, out);
}
